// IPABindingPredictor_78795470013155
// MI455X (gfx1250) — compile-verified, matrix operand order checked by reading, not run on hardware
//
#include <hip/hip_runtime.h>
#include <math.h>

typedef __attribute__((ext_vector_type(16))) _Float16 v16h;
typedef __attribute__((ext_vector_type(8)))  float    v8f;

#define NRES 512
#define CS   384
#define CZ   128
#define CH   16
#define NH   12
#define PQn  4
#define PVn  8
#define NBLK 12
#define CATW 2112   // NH*(CH + PV*4 + CZ)
#define PROJW 1152  // 192 + 384 + 144 + 432

union PkU { _Float16 h[2]; unsigned u; };

// ---------------------------------------------------------------------------
// Generic batched GEMM: D[bz] = act(A[bz] @ B[bz] + bias) (+ D if accum)
//
// Block: 128 threads (4 waves). Block tile 32(M) x 64(N), K-step 32.
// Operand tiles are cooperatively loaded (branch-free: index clamped + value
// masked), converted to f16 and stored pair-packed (b32) into LDS already in
// the WMMA fragment layout (ISA 7.12.2), so the inner loop is just
// ds_load_b128 fragments + v_wmma_f32_16x16x32_f16.
//
// Fragment element mapping for 16-bit operands (wave32):
//   k = 16*(v>=4) + 8*lane_half + 2*(v&3) + j      (v = half-pair 0..7, j=0..1)
// A fragment: lane m = lane&15. B fragment: lane n = lane&15.
// C/D tile:   VGPR r -> m = r + 8*(lane>>4), n = lane&15.
//
// K must be a multiple of 32 (all call sites: 128 / 384 / 512 / 2112).
// act: 0 none, 1 relu, 2 silu
// ---------------------------------------------------------------------------
__global__ void wmma_gemm_kernel(const float* __restrict__ A, long lda, long sA,
                                 const float* __restrict__ B, long ldb, long sB,
                                 const float* __restrict__ bias,
                                 float* __restrict__ D, long ldd, long sD,
                                 int M, int N, int K, int act, int accum) {
  __shared__ v16h lb[4][32];   // B fragments: [wave n-subtile][lane]
  __shared__ v16h la[2][32];   // A fragments: [m subtile][lane]
  unsigned* lbu = (unsigned*)lb;    // pair-packed view: index = frag*8 + v
  unsigned* lau = (unsigned*)la;

  int t    = threadIdx.x;
  int lane = t & 31;
  int wave = t >> 5;
  int n0 = blockIdx.x * 64;
  int m0 = blockIdx.y * 32;

  const float* Ab = A + (long)blockIdx.z * sA;
  const float* Bb = B + (long)blockIdx.z * sB;
  float*       Db = D + (long)blockIdx.z * sD;

  // ---- B staging map: thread owns col n = t&63, k half (t>>6)*16 + pairs --
  int bn    = t & 63;
  int bw    = bn >> 4;                 // destination wave fragment
  int bnl   = bn & 15;                 // lane low bits
  int gn    = n0 + bn;
  int gnc   = gn < N ? gn : (N - 1);   // clamped (always in-bounds)
  float bmsk = gn < N ? 1.f : 0.f;
  int khalf = t >> 6;                  // 0/1 -> k base 0/16

  // ---- A staging map: thread owns row r = t>>2, k octet q = t&3 ----------
  int ar   = t >> 2;                   // 0..31
  int amt  = ar >> 4;                  // m subtile
  int aml  = ar & 15;
  int q    = t & 3;
  int gm   = m0 + ar;
  int gmc  = gm < M ? gm : (M - 1);
  float amsk = gm < M ? 1.f : 0.f;

  v8f c0 = {0.f, 0.f, 0.f, 0.f, 0.f, 0.f, 0.f, 0.f};
  v8f c1 = {0.f, 0.f, 0.f, 0.f, 0.f, 0.f, 0.f, 0.f};

  for (int k0 = 0; k0 < K; k0 += 32) {
    const float* bcol = Bb + (long)k0 * ldb + gnc;
    const float* arow = Ab + (long)gmc * lda + k0 + (q << 3);

    // prefetch next K-step (speculative; past-the-end is dropped, ISA 10.5)
    __builtin_prefetch(bcol + 32 * (long)ldb, 0, 0);
    __builtin_prefetch(arow + 32, 0, 0);

    // ---- stage B tile (32 x 64): 8 k-pairs per thread, packed b32 stores --
#pragma unroll
    for (int i = 0; i < 8; ++i) {
      int k = (khalf << 4) + (i << 1);
      float v0 = bcol[(long)k * ldb] * bmsk;
      float v1 = bcol[(long)(k + 1) * ldb] * bmsk;
      int l  = bnl | (((i >> 2) & 1) << 4);
      int vv = (khalf << 2) + (i & 3);
      PkU p; p.h[0] = (_Float16)v0; p.h[1] = (_Float16)v1;
      lbu[(((bw << 5) | l) << 3) + vv] = p.u;
    }
    // ---- stage A tile (32 x 32): 4 k-pairs per thread (8 contiguous f32) --
#pragma unroll
    for (int j = 0; j < 4; ++j) {
      float v0 = arow[(j << 1)] * amsk;
      float v1 = arow[(j << 1) + 1] * amsk;
      int l  = aml | ((q & 1) << 4);
      int vv = ((q >> 1) << 2) + j;
      PkU p; p.h[0] = (_Float16)v0; p.h[1] = (_Float16)v1;
      lau[(((amt << 5) | l) << 3) + vv] = p.u;
    }
    __syncthreads();
    v16h bf = lb[wave][lane];
    v16h a0 = la[0][lane];
    v16h a1 = la[1][lane];
    c0 = __builtin_amdgcn_wmma_f32_16x16x32_f16(false, a0, false, bf, (short)0,
                                                c0, false, false);
    c1 = __builtin_amdgcn_wmma_f32_16x16x32_f16(false, a1, false, bf, (short)0,
                                                c1, false, false);
    __syncthreads();
  }

  // ---- epilogue ----
  int nn = n0 + (wave << 4) + (lane & 15);
  int mo = (lane >> 4) << 3;
  if (nn < N) {
    float bval = bias ? bias[nn] : 0.0f;
#pragma unroll
    for (int half = 0; half < 2; ++half) {
      v8f cc = half ? c1 : c0;
#pragma unroll
      for (int r = 0; r < 8; ++r) {
        int m = m0 + (half << 4) + r + mo;
        if (m < M) {
          float val = cc[r] + bval;
          if (act == 1)      val = fmaxf(val, 0.0f);
          else if (act == 2) val = val / (1.0f + __expf(-val));
          long idx = (long)m * ldd + nn;
          if (accum) val += Db[idx];
          Db[idx] = val;
        }
      }
    }
  }
}

// ---------------------------------------------------------------------------
// LayerNorm: one block (128 thr) per row; in-place safe (same-thread columns).
// ---------------------------------------------------------------------------
__global__ void ln_kernel(const float* __restrict__ x, const float* __restrict__ g,
                          const float* __restrict__ b, float* __restrict__ y,
                          int cols) {
  __shared__ float red[128];
  long row = blockIdx.x;
  const float* xr = x + row * cols;
  float*       yr = y + row * cols;
  int t = threadIdx.x;

  float s = 0.f;
  for (int c = t; c < cols; c += 128) s += xr[c];
  red[t] = s; __syncthreads();
  for (int o = 64; o > 0; o >>= 1) { if (t < o) red[t] += red[t + o]; __syncthreads(); }
  float mean = red[0] / (float)cols; __syncthreads();

  float v = 0.f;
  for (int c = t; c < cols; c += 128) { float d = xr[c] - mean; v += d * d; }
  red[t] = v; __syncthreads();
  for (int o = 64; o > 0; o >>= 1) { if (t < o) red[t] += red[t + o]; __syncthreads(); }
  float rstd = rsqrtf(red[0] / (float)cols + 1e-5f);
  __syncthreads();

  for (int c = t; c < cols; c += 128)
    yr[c] = (xr[c] - mean) * rstd * g[c] + b[c];
}

// ---------------------------------------------------------------------------
// Pack [Wq|Wkv|Wqp|Wkvp] -> Wcat (384 x 1152), biases -> bcat.
// ---------------------------------------------------------------------------
__global__ void pack_w_kernel(const float* __restrict__ Wq, const float* __restrict__ Wkv,
                              const float* __restrict__ Wqp, const float* __restrict__ Wkvp,
                              const float* __restrict__ bq, const float* __restrict__ bkv,
                              const float* __restrict__ bqp, const float* __restrict__ bkvp,
                              float* __restrict__ Wcat, float* __restrict__ bcat) {
  int idx = blockIdx.x * 256 + threadIdx.x;
  if (idx >= CS * PROJW) return;
  int r = idx / PROJW, c = idx % PROJW;
  float wv, bv;
  if (c < 192)      { wv = Wq[r * 192 + c];          bv = bq[c]; }
  else if (c < 576) { wv = Wkv[r * 384 + (c - 192)]; bv = bkv[c - 192]; }
  else if (c < 720) { wv = Wqp[r * 144 + (c - 576)]; bv = bqp[c - 576]; }
  else              { wv = Wkvp[r * 432 + (c - 720)];bv = bkvp[c - 720]; }
  Wcat[idx] = wv;
  if (r == 0) bcat[c] = bv;
}

// ---------------------------------------------------------------------------
// Init rigid frames: quat=(1,0,0,0), trans=0, rot=I
// ---------------------------------------------------------------------------
__global__ void init_frames_kernel(float* __restrict__ quat, float* __restrict__ trans,
                                   float* __restrict__ rot) {
  int i = blockIdx.x * 128 + threadIdx.x;
  if (i >= NRES) return;
  quat[i * 4 + 0] = 1.f; quat[i * 4 + 1] = 0.f;
  quat[i * 4 + 2] = 0.f; quat[i * 4 + 3] = 0.f;
  trans[i * 3 + 0] = 0.f; trans[i * 3 + 1] = 0.f; trans[i * 3 + 2] = 0.f;
  float* R = rot + i * 9;
  R[0] = 1.f; R[1] = 0.f; R[2] = 0.f;
  R[3] = 0.f; R[4] = 1.f; R[5] = 0.f;
  R[6] = 0.f; R[7] = 0.f; R[8] = 1.f;
}

// ---------------------------------------------------------------------------
// Transform projected points to global frame:
//   q_pts (i,48,3), kv_pts (i,144,3) from proj cols [576,720) and [720,1152).
// ---------------------------------------------------------------------------
__global__ void pts_kernel(const float* __restrict__ proj, const float* __restrict__ rot,
                           const float* __restrict__ trans, float* __restrict__ qpts,
                           float* __restrict__ kvpts) {
  int i = blockIdx.x;
  int t = threadIdx.x;
  if (t >= 144) return;
  const float* R   = rot + i * 9;
  const float* T   = trans + i * 3;
  const float* row = proj + (long)i * PROJW;

  float x = row[720 + 0 * 144 + t];
  float y = row[720 + 1 * 144 + t];
  float z = row[720 + 2 * 144 + t];
  float* kp = kvpts + ((long)i * 144 + t) * 3;
  kp[0] = R[0] * x + R[1] * y + R[2] * z + T[0];
  kp[1] = R[3] * x + R[4] * y + R[5] * z + T[1];
  kp[2] = R[6] * x + R[7] * y + R[8] * z + T[2];

  if (t < 48) {
    float qx = row[576 + 0 * 48 + t];
    float qy = row[576 + 1 * 48 + t];
    float qz = row[576 + 2 * 48 + t];
    float* qp = qpts + ((long)i * 48 + t) * 3;
    qp[0] = R[0] * qx + R[1] * qy + R[2] * qz + T[0];
    qp[1] = R[3] * qx + R[4] * qy + R[5] * qz + T[1];
    qp[2] = R[6] * qx + R[7] * qy + R[8] * qz + T[2];
  }
}

// ---------------------------------------------------------------------------
// Fused attention logits + softmax. One block per (i, h); 256 thr, 2 keys each.
// ---------------------------------------------------------------------------
__global__ void attn_kernel(const float* __restrict__ proj,
                            const float* __restrict__ qpts,
                            const float* __restrict__ kvpts,
                            const float* __restrict__ biasb,
                            const float* __restrict__ head_w,
                            const float* __restrict__ mask,
                            float* __restrict__ att) {
  int i = blockIdx.x;
  int h = blockIdx.y;
  int t = threadIdx.x;
  __shared__ float qv[CH];
  __shared__ float qp[PQn * 3];
  __shared__ float red[256];

  if (t < CH) qv[t] = proj[(long)i * PROJW + h * 16 + t];
  if (t < PQn * 3)
    qp[t] = qpts[((long)i * 48 + h * 4 + (t / 3)) * 3 + (t % 3)];
  __syncthreads();

  float hw = logf(1.0f + __expf(head_w[h])) * 0.13608276348795434f;  // sqrt(1/54)
  float mi = mask[i];

  float l[2];
#pragma unroll
  for (int jj = 0; jj < 2; ++jj) {
    int j = t + jj * 256;
    const float* krow = proj + (long)j * PROJW + 192 + h * 32;
    float dot = 0.f;
#pragma unroll
    for (int c = 0; c < CH; ++c) dot += qv[c] * krow[c];
    float pt = 0.f;
#pragma unroll
    for (int p = 0; p < PQn; ++p) {
      const float* kp = kvpts + ((long)j * 144 + h * 12 + p) * 3;
      float dx = qp[p * 3 + 0] - kp[0];
      float dy = qp[p * 3 + 1] - kp[1];
      float dz = qp[p * 3 + 2] - kp[2];
      pt += dx * dx + dy * dy + dz * dz;
    }
    l[jj] = dot * 0.14433756729740643f                                // sqrt(1/48)
          + 0.5773502691896258f * biasb[((long)i * NRES + j) * NH + h]
          - 0.5f * hw * pt
          + 100000.0f * (mi * mask[j] - 1.0f);
  }

  red[t] = fmaxf(l[0], l[1]); __syncthreads();
  for (int o = 128; o > 0; o >>= 1) { if (t < o) red[t] = fmaxf(red[t], red[t + o]); __syncthreads(); }
  float gmax = red[0]; __syncthreads();

  float e0 = __expf(l[0] - gmax), e1 = __expf(l[1] - gmax);
  red[t] = e0 + e1; __syncthreads();
  for (int o = 128; o > 0; o >>= 1) { if (t < o) red[t] += red[t + o]; __syncthreads(); }
  float inv = 1.0f / red[0];

  float* arow = att + (long)h * NRES * NRES + (long)i * NRES;
  arow[t]       = e0 * inv;
  arow[t + 256] = e1 * inv;
}

// ---------------------------------------------------------------------------
// o_pt back-rotation (R^T (o_pt - t)) + norms -> cat columns [192, 576)
// ---------------------------------------------------------------------------
__global__ void post_pt_kernel(const float* __restrict__ opt, const float* __restrict__ rot,
                               const float* __restrict__ trans, float* __restrict__ cat) {
  int i = blockIdx.x;
  int t = threadIdx.x;            // h*8 + p
  if (t >= NH * PVn) return;
  int h = t >> 3, p = t & 7;
  const float* o = opt + (long)i * 288 + h * 24 + p * 3;
  const float* R = rot + i * 9;
  const float* T = trans + i * 3;
  float x = o[0] - T[0], y = o[1] - T[1], z = o[2] - T[2];
  float rx = R[0] * x + R[3] * y + R[6] * z;
  float ry = R[1] * x + R[4] * y + R[7] * z;
  float rz = R[2] * x + R[5] * y + R[8] * z;
  float nrm = sqrtf(rx * rx + ry * ry + rz * rz + 1e-8f);
  float* crow = cat + (long)i * CATW;
  crow[192 + 0 * 96 + t] = rx;
  crow[192 + 1 * 96 + t] = ry;
  crow[192 + 2 * 96 + t] = rz;
  crow[480 + t]          = nrm;
}

// ---------------------------------------------------------------------------
// Backbone update: upd = s @ Wbb + bbb; quaternion composition + renorm;
// trans += R_old @ tv; rot = R(new quat). One thread per residue.
// ---------------------------------------------------------------------------
__global__ void backbone_kernel(const float* __restrict__ s, const float* __restrict__ Wbb,
                                const float* __restrict__ bbb, float* __restrict__ quat,
                                float* __restrict__ trans, float* __restrict__ rot) {
  int i = blockIdx.x * 128 + threadIdx.x;
  if (i >= NRES) return;
  float u[6];
#pragma unroll
  for (int d = 0; d < 6; ++d) u[d] = bbb[d];
  const float* sr = s + (long)i * CS;
  for (int c = 0; c < CS; ++c) {
    float sv = sr[c];
    const float* wr = Wbb + c * 6;
#pragma unroll
    for (int d = 0; d < 6; ++d) u[d] += sv * wr[d];
  }
  float w_ = quat[i * 4 + 0], x_ = quat[i * 4 + 1];
  float y_ = quat[i * 4 + 2], z_ = quat[i * 4 + 3];
  float vx = u[0], vy = u[1], vz = u[2];
  float nw = w_ + (-x_ * vx - y_ * vy - z_ * vz);
  float nx = x_ + ( w_ * vx + y_ * vz - z_ * vy);
  float ny = y_ + ( w_ * vy - x_ * vz + z_ * vx);
  float nz = z_ + ( w_ * vz + x_ * vy - y_ * vx);
  float inv = rsqrtf(nw * nw + nx * nx + ny * ny + nz * nz);
  nw *= inv; nx *= inv; ny *= inv; nz *= inv;

  const float* R = rot + i * 9;
  float tx = R[0] * u[3] + R[1] * u[4] + R[2] * u[5] + trans[i * 3 + 0];
  float ty = R[3] * u[3] + R[4] * u[4] + R[5] * u[5] + trans[i * 3 + 1];
  float tz = R[6] * u[3] + R[7] * u[4] + R[8] * u[5] + trans[i * 3 + 2];

  quat[i * 4 + 0] = nw; quat[i * 4 + 1] = nx;
  quat[i * 4 + 2] = ny; quat[i * 4 + 3] = nz;
  trans[i * 3 + 0] = tx; trans[i * 3 + 1] = ty; trans[i * 3 + 2] = tz;

  float* Ro = rot + i * 9;
  Ro[0] = 1.f - 2.f * (ny * ny + nz * nz); Ro[1] = 2.f * (nx * ny - nw * nz); Ro[2] = 2.f * (nx * nz + nw * ny);
  Ro[3] = 2.f * (nx * ny + nw * nz); Ro[4] = 1.f - 2.f * (nx * nx + nz * nz); Ro[5] = 2.f * (ny * nz - nw * nx);
  Ro[6] = 2.f * (nx * nz - nw * ny); Ro[7] = 2.f * (ny * nz + nw * nx); Ro[8] = 1.f - 2.f * (nx * nx + ny * ny);
}

// ---------------------------------------------------------------------------
// Final: out[0] = sum_i (h[i,:] . Wa2 + ba2)
// ---------------------------------------------------------------------------
__global__ void head_reduce_kernel(const float* __restrict__ h, const float* __restrict__ Wa2,
                                   const float* __restrict__ ba2, float* __restrict__ out) {
  __shared__ float red[256];
  int t = threadIdx.x;
  float s = 0.f;
  for (int idx = t; idx < NRES * CS; idx += 256)
    s += h[idx] * Wa2[idx % CS];
  red[t] = s; __syncthreads();
  for (int o = 128; o > 0; o >>= 1) { if (t < o) red[t] += red[t + o]; __syncthreads(); }
  if (t == 0) out[0] = red[0] + (float)NRES * ba2[0];
}

// ---------------------------------------------------------------------------
// Host orchestration
// ---------------------------------------------------------------------------
extern "C" void kernel_launch(void* const* d_in, const int* in_sizes, int n_in,
                              void* d_out, int out_size, void* d_ws, size_t ws_size,
                              hipStream_t stream) {
  const float* s_in  = (const float*)d_in[0];
  const float* z_in  = (const float*)d_in[1];
  const float* nmask = (const float*)d_in[2];
  const float* ln_s_g = (const float*)d_in[3],  * ln_s_b = (const float*)d_in[4];
  const float* ln_z_g = (const float*)d_in[5],  * ln_z_b = (const float*)d_in[6];
  const float* W_in   = (const float*)d_in[7],  * b_in   = (const float*)d_in[8];
  const float* Wq     = (const float*)d_in[9],  * bq     = (const float*)d_in[10];
  const float* Wkv    = (const float*)d_in[11], * bkv    = (const float*)d_in[12];
  const float* Wqp    = (const float*)d_in[13], * bqp    = (const float*)d_in[14];
  const float* Wkvp   = (const float*)d_in[15], * bkvp   = (const float*)d_in[16];
  const float* Wb     = (const float*)d_in[17], * bb     = (const float*)d_in[18];
  const float* head_w = (const float*)d_in[19];
  const float* Wout   = (const float*)d_in[20], * bout   = (const float*)d_in[21];
  const float* ln_i_g = (const float*)d_in[22], * ln_i_b = (const float*)d_in[23];
  const float* Wt1 = (const float*)d_in[24], * bt1 = (const float*)d_in[25];
  const float* Wt2 = (const float*)d_in[26], * bt2 = (const float*)d_in[27];
  const float* Wt3 = (const float*)d_in[28], * bt3 = (const float*)d_in[29];
  const float* ln_t_g = (const float*)d_in[30], * ln_t_b = (const float*)d_in[31];
  const float* Wbb = (const float*)d_in[32], * bbb = (const float*)d_in[33];
  const float* Wa1 = (const float*)d_in[34], * ba1 = (const float*)d_in[35];
  const float* Wa2 = (const float*)d_in[36], * ba2 = (const float*)d_in[37];
  float* out = (float*)d_out;

  float* ws = (float*)d_ws;
  size_t off = 0;
  float* ZLN   = ws + off; off += (size_t)NRES * NRES * CZ;   // 33.5M floats
  float* BIAS  = ws + off; off += (size_t)NRES * NRES * NH;   // 3.1M
  float* ATT   = ws + off; off += (size_t)NH * NRES * NRES;   // 3.1M
  float* SCUR  = ws + off; off += (size_t)NRES * CS;
  float* T1    = ws + off; off += (size_t)NRES * CS;
  float* T2    = ws + off; off += (size_t)NRES * CS;
  float* PROJ  = ws + off; off += (size_t)NRES * PROJW;
  float* QPTS  = ws + off; off += (size_t)NRES * 48 * 3;
  float* KVPTS = ws + off; off += (size_t)NRES * 144 * 3;
  float* OPT   = ws + off; off += (size_t)NRES * 288;
  float* CAT   = ws + off; off += (size_t)NRES * CATW;
  float* WCAT  = ws + off; off += (size_t)CS * PROJW;
  float* BCAT  = ws + off; off += (size_t)PROJW;
  float* QUAT  = ws + off; off += (size_t)NRES * 4;
  float* TRANS = ws + off; off += (size_t)NRES * 3;
  float* ROT   = ws + off; off += (size_t)NRES * 9;
  float* HBUF  = ws + off; off += (size_t)NRES * CS;
  (void)in_sizes; (void)n_in; (void)out_size; (void)ws_size;

  auto gemm = [&](const float* A, long lda, long sA, const float* B, long ldb, long sB,
                  const float* bias, float* D, long ldd, long sD,
                  int M, int N, int K, int batch, int act, int accum) {
    dim3 grid((N + 63) / 64, (M + 31) / 32, batch);
    wmma_gemm_kernel<<<grid, dim3(128), 0, stream>>>(A, lda, sA, B, ldb, sB, bias,
                                                     D, ldd, sD, M, N, K, act, accum);
  };

  // --- One-time (per call) setup -------------------------------------------
  ln_kernel<<<NRES, 128, 0, stream>>>(s_in, ln_s_g, ln_s_b, T1, CS);
  gemm(T1, CS, 0, W_in, CS, 0, b_in, SCUR, CS, 0, NRES, CS, CS, 1, 0, 0);

  ln_kernel<<<NRES * NRES, 128, 0, stream>>>(z_in, ln_z_g, ln_z_b, ZLN, CZ);
  // Hoisted pair bias (loop-invariant in the reference): (N*N, 128) @ (128, 12)
  gemm(ZLN, CZ, 0, Wb, NH, 0, bb, BIAS, NH, 0, NRES * NRES, NH, CZ, 1, 0, 0);

  pack_w_kernel<<<(CS * PROJW + 255) / 256, 256, 0, stream>>>(
      Wq, Wkv, Wqp, Wkvp, bq, bkv, bqp, bkvp, WCAT, BCAT);
  init_frames_kernel<<<(NRES + 127) / 128, 128, 0, stream>>>(QUAT, TRANS, ROT);

  // --- 12 IPA + transition blocks ------------------------------------------
  for (int blk = 0; blk < NBLK; ++blk) {
    // fused q|kv|qp|kvp projection: (512,384) @ (384,1152)
    gemm(SCUR, CS, 0, WCAT, PROJW, 0, BCAT, PROJ, PROJW, 0, NRES, PROJW, CS, 1, 0, 0);

    pts_kernel<<<NRES, 160, 0, stream>>>(PROJ, ROT, TRANS, QPTS, KVPTS);

    attn_kernel<<<dim3(NRES, NH), 256, 0, stream>>>(PROJ, QPTS, KVPTS, BIAS,
                                                    head_w, nmask, ATT);

    // o[i, h*16+c] = sum_j a[h,i,j] v[j,h,c]   (batched over heads)
    gemm(ATT, NRES, (long)NRES * NRES, PROJ + 208, PROJW, 32, nullptr,
         CAT, CATW, 16, NRES, CH, NRES, NH, 0, 0);
    // o_pt raw: (i, h*24 + p*3 + c)
    gemm(ATT, NRES, (long)NRES * NRES, KVPTS + 12, 432, 36, nullptr,
         OPT, 288, 24, NRES, PVn * 3, NRES, NH, 0, 0);

    post_pt_kernel<<<NRES, 96, 0, stream>>>(OPT, ROT, TRANS, CAT);

    // o_pair: per-query-row GEMM (12 x 128, K=512), batched over i=512.
    gemm(ATT, (long)NRES * NRES, NRES, ZLN, CZ, (long)NRES * CZ, nullptr,
         CAT + 576, CZ, CATW, NH, CZ, NRES, NRES, 0, 0);

    // out projection + residual: s += cat @ Wout + bout
    gemm(CAT, CATW, 0, Wout, CS, 0, bout, SCUR, CS, 0, NRES, CS, CATW, 1, 0, 1);
    ln_kernel<<<NRES, 128, 0, stream>>>(SCUR, ln_i_g, ln_i_b, SCUR, CS);

    // transition MLP
    gemm(SCUR, CS, 0, Wt1, CS, 0, bt1, T1, CS, 0, NRES, CS, CS, 1, 1, 0);
    gemm(T1,   CS, 0, Wt2, CS, 0, bt2, T2, CS, 0, NRES, CS, CS, 1, 1, 0);
    gemm(T2,   CS, 0, Wt3, CS, 0, bt3, SCUR, CS, 0, NRES, CS, CS, 1, 0, 1);
    ln_kernel<<<NRES, 128, 0, stream>>>(SCUR, ln_t_g, ln_t_b, SCUR, CS);

    backbone_kernel<<<(NRES + 127) / 128, 128, 0, stream>>>(SCUR, Wbb, bbb,
                                                            QUAT, TRANS, ROT);
  }

  // --- Head -----------------------------------------------------------------
  gemm(SCUR, CS, 0, Wa1, CS, 0, ba1, HBUF, CS, 0, NRES, CS, CS, 1, 2, 0); // silu
  head_reduce_kernel<<<1, 256, 0, stream>>>(HBUF, Wa2, ba2, out);
}
